// GAT_76957224010100
// MI455X (gfx1250) — matrix operand order repaired and checked by reading, not compiled, not run on hardware
//
#include <hip/hip_runtime.h>
#include <stdint.h>

#define N_NODES 50000
#define N_EDGES 800000
#define HID     128
#define NHEAD   4
#define FOUT    32

typedef __attribute__((ext_vector_type(16))) __bf16 v16bf;
typedef __attribute__((ext_vector_type(8)))  float  v8f;

// ---- helpers -------------------------------------------------------------

__device__ __forceinline__ __bf16 f2bf(float f) {
    // round-to-nearest-even fp32 -> bf16
    union { float f; uint32_t u; } v; v.f = f;
    uint32_t r = (v.u + 0x7FFFu + ((v.u >> 16) & 1u)) >> 16;
    uint16_t hs = (uint16_t)r;
    __bf16 b;
    __builtin_memcpy(&b, &hs, 2);
    return b;
}

// monotonic float<->uint mapping for atomic max on possibly-negative floats
__device__ __forceinline__ uint32_t enc_max(float f) {
    uint32_t u = __float_as_uint(f);
    return (u & 0x80000000u) ? ~u : (u | 0x80000000u);
}
__device__ __forceinline__ float dec_max(uint32_t k) {
    return __uint_as_float((k & 0x80000000u) ? (k & 0x7FFFFFFFu) : ~k);
}

__device__ __forceinline__ void fatomic_add(float* p, float v) {
    __hip_atomic_fetch_add(p, v, __ATOMIC_RELAXED, __HIP_MEMORY_SCOPE_AGENT);
}

// ---- kernels -------------------------------------------------------------

__global__ void k_zero_u32(uint32_t* __restrict__ p, int n) {
    int t = blockIdx.x * blockDim.x + threadIdx.x;
    if (t < n) p[t] = 0u;
}

// C[M,128] = X[M,KC] @ W[KC,128] via bf16 WMMA, fp32 accumulate.
// One block (256 thr = 8 waves) per 16-row strip; wave w computes N-tile w.
// W is staged once per block into LDS as bf16, column-major (K padded to 32),
// so each B fragment is two aligned 16-byte LDS reads per lane.
template <int KC>
__global__ void __launch_bounds__(256)
k_gemm_wmma(const float* __restrict__ X,
            const float* __restrict__ W,
            float* __restrict__ C) {
    constexpr int KP = (KC + 31) & ~31;       // K padded to multiple of 32
    __shared__ __bf16 Wl[HID * KP];           // [col][k] column-major, bf16

    const int tid = threadIdx.x;

    // cooperative global->LDS convert of W (zero-pad the K tail)
    #pragma unroll 4
    for (int idx = tid; idx < KP * HID; idx += 256) {
        int k = idx >> 7;          // 0..KP-1
        int c = idx & (HID - 1);   // 0..127
        float v = (k < KC) ? W[k * HID + c] : 0.0f;
        Wl[c * KP + k] = f2bf(v);
    }
    __syncthreads();

    const int nt   = tid >> 5;       // wave id = N-tile 0..7
    const int lane = tid & 31;
    const int mt   = blockIdx.x;     // 0..3124
    const int r    = lane & 15;
    const int half = lane >> 4;

    const float* xrow = X + (size_t)(mt * 16 + r) * KC;
    const __bf16* wcol = &Wl[(nt * 16 + r) * KP];

    v8f acc = {};
    #pragma unroll
    for (int k0 = 0; k0 < KP; k0 += 32) {
        // A fragment (16x32 bf16): half=0 -> {K0..7, K16..23}; half=1 -> {K8..15, K24..31}
        v16bf a;
        const int ka0 = k0 + half * 8;
        #pragma unroll
        for (int e = 0; e < 8; ++e) {
            int kl = ka0 + e, kh = ka0 + 16 + e;
            a[e]     = f2bf((kl < KC) ? xrow[kl] : 0.0f);
            a[8 + e] = f2bf((kh < KC) ? xrow[kh] : 0.0f);
        }
        // B fragment (32x16 bf16): half=0 -> {K0..7, K16..23}; half=1 -> {K8..15, K24..31},
        // column = nt*16 + (lane&15). Two 16B LDS reads (16B aligned).
        typedef __attribute__((ext_vector_type(8))) __bf16 v8bf;
        v8bf b0 = *(const v8bf*)(wcol + k0 + half * 8);
        v8bf b1 = *(const v8bf*)(wcol + k0 + 16 + half * 8);
        v16bf b = __builtin_shufflevector(b0, b1, 0,1,2,3,4,5,6,7,8,9,10,11,12,13,14,15);
        acc = __builtin_amdgcn_wmma_f32_16x16x32_bf16(
            false, a, false, b, (short)0, acc, false, false);
    }
    // C/D layout: VGPR v -> row v + half*8, col = lane&15
    #pragma unroll
    for (int v = 0; v < 8; ++v) {
        C[(size_t)(mt * 16 + v + half * 8) * HID + nt * 16 + r] = acc[v];
    }
}

// el[n,h] = dot(ft[n,h,:], al[h,:]); er likewise.
__global__ void k_attn(const float* __restrict__ ft,
                       const float* __restrict__ al,
                       const float* __restrict__ ar,
                       float* __restrict__ el, float* __restrict__ er) {
    int tid = blockIdx.x * blockDim.x + threadIdx.x;
    if (tid >= N_NODES * NHEAD) return;
    int n = tid >> 2, h = tid & 3;
    const float* f = ft + (size_t)n * HID + h * FOUT;
    float sl = 0.f, sr = 0.f;
    #pragma unroll 8
    for (int i = 0; i < FOUT; ++i) {
        float v = f[i];
        sl += v * al[h * FOUT + i];
        sr += v * ar[h * FOUT + i];
    }
    el[tid] = sl;
    er[tid] = sr;
}

// pass 1: leaky-relu logit per (edge,head); atomic-max into m (encoded)
__global__ void k_edge_logit(const float* __restrict__ el,
                             const float* __restrict__ er,
                             const int* __restrict__ src,
                             const int* __restrict__ dst,
                             float* __restrict__ ebuf,
                             uint32_t* __restrict__ menc) {
    int tid = blockIdx.x * blockDim.x + threadIdx.x;
    if (tid >= N_EDGES * NHEAD) return;
    int e = tid >> 2, h = tid & 3;
    int s = src[e], d = dst[e];
    float v = el[s * NHEAD + h] + er[d * NHEAD + h];
    v = v > 0.f ? v : 0.2f * v;               // leaky relu, slope 0.2
    ebuf[tid] = v;
    atomicMax(&menc[d * NHEAD + h], enc_max(v));
}

// pass 2: ex = exp(e - m[dst]); atomic-add into s
__global__ void k_edge_expsum(float* __restrict__ ebuf,
                              const uint32_t* __restrict__ menc,
                              const int* __restrict__ dst,
                              float* __restrict__ sbuf) {
    int tid = blockIdx.x * blockDim.x + threadIdx.x;
    if (tid >= N_EDGES * NHEAD) return;
    int e = tid >> 2, h = tid & 3;
    int d = dst[e];
    float m  = dec_max(menc[d * NHEAD + h]);
    float ex = __expf(ebuf[tid] - m);
    ebuf[tid] = ex;
    fatomic_add(&sbuf[d * NHEAD + h], ex);
}

// pass 3: 32 lanes per edge; lane g moves floats [4g,4g+4) of ft[src], scaled
// by alpha of head g/8, atomically accumulated into rst[dst].
__global__ void k_edge_scatter(const float* __restrict__ ebuf,
                               const float* __restrict__ sbuf,
                               const float* __restrict__ ft,
                               const int* __restrict__ src,
                               const int* __restrict__ dst,
                               float* __restrict__ rst) {
    int tid = blockIdx.x * blockDim.x + threadIdx.x;
    if (tid >= N_EDGES * 32) return;
    int e = tid >> 5, g = tid & 31;
    int s = src[e], d = dst[e];
    int h = g >> 3;
    float alpha = ebuf[e * NHEAD + h] / sbuf[d * NHEAD + h];
    const float4 v = *(const float4*)(ft + (size_t)s * HID + g * 4);
    float* o = rst + (size_t)d * HID + g * 4;
    fatomic_add(o + 0, alpha * v.x);
    fatomic_add(o + 1, alpha * v.y);
    fatomic_add(o + 2, alpha * v.z);
    fatomic_add(o + 3, alpha * v.w);
}

__global__ void k_bias(float* __restrict__ hbuf, const float* __restrict__ bias) {
    int tid = blockIdx.x * blockDim.x + threadIdx.x;
    if (tid >= N_NODES * HID) return;
    hbuf[tid] += bias[tid & (HID - 1)];
}

// one wave per node: mean over 128 features
__global__ void k_mean(const float* __restrict__ hbuf, float* __restrict__ out) {
    int wave = (blockIdx.x * blockDim.x + threadIdx.x) >> 5;
    int lane = threadIdx.x & 31;
    if (wave >= N_NODES) return;
    const float* p = hbuf + (size_t)wave * HID;
    float s = p[lane] + p[lane + 32] + p[lane + 64] + p[lane + 96];
    #pragma unroll
    for (int off = 16; off > 0; off >>= 1) s += __shfl_down(s, off, 32);
    if (lane == 0) out[wave] = s * (1.0f / (float)HID);
}

// ---- launch --------------------------------------------------------------

template <int KC>
static void run_layer(const float* x,
                      const float* Wt, const float* al, const float* ar,
                      const float* bias,
                      const int* src, const int* dst,
                      float* ft, float* hbuf, float* el, float* er,
                      uint32_t* menc, float* sbuf, float* ebuf,
                      hipStream_t stream) {
    // GEMM: ft = x @ W  (one block per 16-row strip, 8 waves = 8 N-tiles)
    k_gemm_wmma<KC><<<N_NODES / 16, 256, 0, stream>>>(x, Wt, ft);

    // attention coefficients per node
    k_attn<<<(N_NODES * NHEAD + 255) / 256, 256, 0, stream>>>(ft, al, ar, el, er);

    // zero softmax state (menc,sbuf contiguous) and accumulator
    k_zero_u32<<<(N_NODES * NHEAD * 2 + 255) / 256, 256, 0, stream>>>(menc, N_NODES * NHEAD * 2);
    k_zero_u32<<<(N_NODES * HID + 255) / 256, 256, 0, stream>>>((uint32_t*)hbuf, N_NODES * HID);

    // edge softmax + scatter
    k_edge_logit<<<(N_EDGES * NHEAD + 255) / 256, 256, 0, stream>>>(el, er, src, dst, ebuf, menc);
    k_edge_expsum<<<(N_EDGES * NHEAD + 255) / 256, 256, 0, stream>>>(ebuf, menc, dst, sbuf);
    k_edge_scatter<<<(N_EDGES * 32 + 255) / 256, 256, 0, stream>>>(ebuf, sbuf, ft, src, dst, hbuf);

    k_bias<<<(N_NODES * HID + 255) / 256, 256, 0, stream>>>(hbuf, bias);
}

extern "C" void kernel_launch(void* const* d_in, const int* in_sizes, int n_in,
                              void* d_out, int out_size, void* d_ws, size_t ws_size,
                              hipStream_t stream) {
    const float* features = (const float*)d_in[0];  // [N,165]
    const float* W1   = (const float*)d_in[1];      // [165,128]
    const float* al1  = (const float*)d_in[2];      // [4,32]
    const float* ar1  = (const float*)d_in[3];
    const float* b1   = (const float*)d_in[4];      // [128]
    const float* W2   = (const float*)d_in[5];      // [128,128]
    const float* al2  = (const float*)d_in[6];
    const float* ar2  = (const float*)d_in[7];
    const float* b2   = (const float*)d_in[8];
    const int*   src  = (const int*)d_in[9];        // [E]
    const int*   dst  = (const int*)d_in[10];       // [E]
    float* out = (float*)d_out;                     // [N]

    // workspace carve-up (all fp32/u32, ~68 MB total)
    float*    ws   = (float*)d_ws;
    float*    ft   = ws;                              // N*128
    float*    hbuf = ft + (size_t)N_NODES * HID;      // N*128
    float*    el   = hbuf + (size_t)N_NODES * HID;    // N*4
    float*    er   = el + (size_t)N_NODES * NHEAD;    // N*4
    uint32_t* menc = (uint32_t*)(er + (size_t)N_NODES * NHEAD); // N*4
    float*    sbuf = (float*)(menc + (size_t)N_NODES * NHEAD);  // N*4
    float*    ebuf = sbuf + (size_t)N_NODES * NHEAD;  // E*4

    // layer 1: features (K=165) -> hbuf
    run_layer<165>(features, W1, al1, ar1, b1, src, dst,
                   ft, hbuf, el, er, menc, sbuf, ebuf, stream);
    // layer 2: hbuf (K=128) -> hbuf (GEMM consumes hbuf into ft before hbuf is zeroed)
    run_layer<HID>(hbuf, W2, al2, ar2, b2, src, dst,
                   ft, hbuf, el, er, menc, sbuf, ebuf, stream);

    // final per-node mean over 128 features
    k_mean<<<(N_NODES * 32 + 255) / 256, 256, 0, stream>>>(hbuf, out);
}
